// L2GATConv_84859963834454
// MI455X (gfx1250) — compile-verified, matrix operand order checked by reading, not run on hardware
//
#include <hip/hip_runtime.h>
#include <hip/hip_bf16.h>
#include <cfloat>

// ---------------------------------------------------------------------------
// GATConv x2 for MI455X (gfx1250).
//   Layer1 GEMM (100000x128 @ 128x200): bf16 hi/lo split WMMA
//   (v_wmma_f32_16x16x32_bf16, fp32 accumulate, ~fp32 accuracy).
//   One wave per 16-row M-tile sweeping all 13 N-tiles: x read exactly once,
//   s1/d1 attention projections fused into the epilogue (no h1 re-read),
//   a1 vectors zero-padded so the inner loop stays branch-free (EXEC all-1s).
//   Edge softmax-aggregation: memory/atomic bound (~2.7 GB), float atomics,
//   float4 gathers.
// ---------------------------------------------------------------------------

#define IN_F   128
#define HID    200
#define HID_P  208      // 13 * 16, zero padded for WMMA tiling
#define NTILES 13
#define OUT_F  2
#define NEG_SLOPE 0.2f

typedef __bf16 bf16_t;
typedef __attribute__((ext_vector_type(16))) __bf16 v16bf;
typedef __attribute__((ext_vector_type(8)))  __bf16 v8bf;
typedef __attribute__((ext_vector_type(8)))  float  v8f;

static __device__ __forceinline__ bf16_t bf16_trunc(float f) {
    unsigned u = __float_as_uint(f);
    unsigned short h = (unsigned short)(u >> 16);   // truncate: exact split
    bf16_t b; __builtin_memcpy(&b, &h, 2); return b;
}
static __device__ __forceinline__ float bf16_up(bf16_t b) {
    unsigned short h; __builtin_memcpy(&h, &b, 2);
    return __uint_as_float(((unsigned)h) << 16);
}

static __device__ __forceinline__ void atomicMaxFloat(float* addr, float val) {
    if (val >= 0.0f) atomicMax((int*)addr, __float_as_int(val));
    else             atomicMin((unsigned int*)addr, __float_as_uint(val));
}

static __device__ __forceinline__ float leaky(float t) {
    return t > 0.0f ? t : NEG_SLOPE * t;
}

// ----------------------------- utility kernels -----------------------------

__global__ void fill_f32(float* __restrict__ p, long n, float v) {
    long i = (long)blockIdx.x * blockDim.x + threadIdx.x;
    long stride = (long)gridDim.x * blockDim.x;
    for (; i < n; i += stride) p[i] = v;
}

// Pack W1 (IN_F x HID, row major) into transposed, padded bf16 hi/lo:
// Whi/Wlo[n][k], n in [0,208), k in [0,128), zeros for n >= 200.
// Also zero-pad a1_src/a1_dst to HID_P so the GEMM epilogue is branch-free.
__global__ void pack_w1(const float* __restrict__ W1,
                        const float* __restrict__ a1s, const float* __restrict__ a1d,
                        bf16_t* __restrict__ Whi, bf16_t* __restrict__ Wlo,
                        float* __restrict__ a1s_p, float* __restrict__ a1d_p) {
    int i = blockIdx.x * blockDim.x + threadIdx.x;
    if (i < HID_P) {
        a1s_p[i] = (i < HID) ? a1s[i] : 0.0f;
        a1d_p[i] = (i < HID) ? a1d[i] : 0.0f;
    }
    if (i >= HID_P * IN_F) return;
    int n = i / IN_F, k = i % IN_F;
    float v = (n < HID) ? W1[k * HID + n] : 0.0f;
    bf16_t hi = bf16_trunc(v);
    Whi[i] = hi;
    Wlo[i] = bf16_trunc(v - bf16_up(hi));
}

// ------------------------------- layer-1 GEMM ------------------------------
// One wave per 16-row M-tile. A fragments (4 k-steps, hi/lo) are converted
// once and held in registers while the wave sweeps all 13 N-tiles:
// 13 * 4 * 3 = 156 WMMA issues per wave, x read exactly once from HBM.
// s1/d1 projections accumulated from the D fragments (wave owns full rows).
__global__ __launch_bounds__(256) void gemm1_wmma(
    const float* __restrict__ x, const bf16_t* __restrict__ Whi,
    const bf16_t* __restrict__ Wlo, const float* __restrict__ a1s_p,
    const float* __restrict__ a1d_p, float* __restrict__ h1,
    float* __restrict__ s1, float* __restrict__ d1, int n_nodes)
{
    int wave = (int)((blockIdx.x * blockDim.x + threadIdx.x) >> 5);
    int lane = threadIdx.x & 31;
    int mtiles = n_nodes >> 4;
    if (wave >= mtiles) return;                 // whole-wave exit: EXEC all-1s
    int m0 = wave * 16;
    int sub  = lane >> 4;                       // K-half select
    int lrow = lane & 15;                       // A row / B col / D col

    const float* xrow = x + (long)(m0 + lrow) * IN_F;

    // Load + hi/lo-split all A fragments (K = 128 -> 4 steps of 32).
    v16bf a_hi[4], a_lo[4];
#pragma unroll
    for (int ks = 0; ks < 4; ++ks) {
        int ka = ks * 32 + 8 * sub;             // v16bf elems 0..7  -> K [ka, ka+8)
        int kb = ka + 16;                       // v16bf elems 8..15 -> K [kb, kb+8)
#pragma unroll
        for (int t = 0; t < 8; ++t) {
            float v0 = xrow[ka + t];
            float v1 = xrow[kb + t];
            bf16_t h0 = bf16_trunc(v0);
            bf16_t h1e = bf16_trunc(v1);
            a_hi[ks][t]     = h0;
            a_hi[ks][8 + t] = h1e;
            a_lo[ks][t]     = bf16_trunc(v0 - bf16_up(h0));
            a_lo[ks][8 + t] = bf16_trunc(v1 - bf16_up(h1e));
        }
    }

    float sp[8], dp[8];
#pragma unroll
    for (int r = 0; r < 8; ++r) { sp[r] = 0.f; dp[r] = 0.f; }

    for (int ntile = 0; ntile < NTILES; ++ntile) {
        int n0 = ntile * 16;
        const bf16_t* bh = Whi + (long)(n0 + lrow) * IN_F;
        const bf16_t* bl = Wlo + (long)(n0 + lrow) * IN_F;
        v8f acc = {};
#pragma unroll
        for (int ks = 0; ks < 4; ++ks) {
            int ka = ks * 32 + 8 * sub;
            int kb = ka + 16;
            v8bf b0h = *reinterpret_cast<const v8bf*>(bh + ka);
            v8bf b1h = *reinterpret_cast<const v8bf*>(bh + kb);
            v8bf b0l = *reinterpret_cast<const v8bf*>(bl + ka);
            v8bf b1l = *reinterpret_cast<const v8bf*>(bl + kb);
            v16bf b_hi, b_lo;
#pragma unroll
            for (int t = 0; t < 8; ++t) {
                b_hi[t] = b0h[t]; b_hi[8 + t] = b1h[t];
                b_lo[t] = b0l[t]; b_lo[8 + t] = b1l[t];
            }
            acc = __builtin_amdgcn_wmma_f32_16x16x32_bf16(false, a_lo[ks], false, b_hi, (short)0, acc, false, false);
            acc = __builtin_amdgcn_wmma_f32_16x16x32_bf16(false, a_hi[ks], false, b_lo, (short)0, acc, false, false);
            acc = __builtin_amdgcn_wmma_f32_16x16x32_bf16(false, a_hi[ks], false, b_hi, (short)0, acc, false, false);
        }
        // D layout: VGPR r -> row m0 + r + 8*sub, col n0 + lrow.
        int col = n0 + lrow;
        float as = a1s_p[col];                  // zero-padded: no branch
        float ad = a1d_p[col];
#pragma unroll
        for (int r = 0; r < 8; ++r) {
            h1[(long)(m0 + 8 * sub + r) * HID_P + col] = acc[r];
            sp[r] += acc[r] * as;
            dp[r] += acc[r] * ad;
        }
    }

    // Reduce sp/dp across each 16-lane half (xor masks stay within halves).
#pragma unroll
    for (int r = 0; r < 8; ++r) {
        float s = sp[r], d = dp[r];
#pragma unroll
        for (int o = 8; o > 0; o >>= 1) {
            s += __shfl_xor(s, o, 32);
            d += __shfl_xor(d, o, 32);
        }
        if (lrow == 0) {
            int row = m0 + 8 * sub + r;
            s1[row] = s;
            d1[row] = d;
        }
    }
}

// ------------------------------- edge kernels ------------------------------
// Edge id e in [0, E+N): e < E -> edge_index, else self-loop (e-E, e-E).

__global__ void edge_max(const int* __restrict__ src, const int* __restrict__ dst,
                         int E_, int n_nodes, const float* __restrict__ s,
                         const float* __restrict__ d, float* __restrict__ m)
{
    int e = blockIdx.x * blockDim.x + threadIdx.x;
    int tot = E_ + n_nodes;
    if (e >= tot) return;
    int si = e < E_ ? src[e] : (e - E_);
    int di = e < E_ ? dst[e] : (e - E_);
    atomicMaxFloat(m + di, leaky(s[si] + d[di]));
}

__global__ void edge_expsum(const int* __restrict__ src, const int* __restrict__ dst,
                            int E_, int n_nodes, const float* __restrict__ s,
                            const float* __restrict__ d, const float* __restrict__ m,
                            float* __restrict__ den)
{
    int e = blockIdx.x * blockDim.x + threadIdx.x;
    int tot = E_ + n_nodes;
    if (e >= tot) return;
    int si = e < E_ ? src[e] : (e - E_);
    int di = e < E_ ? dst[e] : (e - E_);
    atomicAdd(den + di, __expf(leaky(s[si] + d[di]) - m[di]));
}

// One wave per edge: gather 200 ch of h1[src] (float4), scatter-add into agg[dst].
__global__ __launch_bounds__(256) void edge_agg1(
    const int* __restrict__ src, const int* __restrict__ dst, int E_, int n_nodes,
    const float* __restrict__ s, const float* __restrict__ d,
    const float* __restrict__ m, const float* __restrict__ den,
    const float* __restrict__ h1, float* __restrict__ agg)
{
    int e = (int)((blockIdx.x * blockDim.x + threadIdx.x) >> 5);
    int lane = threadIdx.x & 31;
    int tot = E_ + n_nodes;
    if (e >= tot) return;
    int si = e < E_ ? src[e] : (e - E_);
    int di = e < E_ ? dst[e] : (e - E_);
    float coef = __expf(leaky(s[si] + d[di]) - m[di]) / (den[di] + 1e-16f);
    const float4* hr4 = reinterpret_cast<const float4*>(h1 + (long)si * HID_P);
    float* ar = agg + (long)di * HID;
#pragma unroll 2
    for (int q = lane; q < HID / 4; q += 32) {   // 50 float4 per edge
        float4 v = hr4[q];
        int f = q * 4;
        atomicAdd(ar + f + 0, coef * v.x);
        atomicAdd(ar + f + 1, coef * v.y);
        atomicAdd(ar + f + 2, coef * v.z);
        atomicAdd(ar + f + 3, coef * v.w);
    }
}

// relu(agg + b1) -> h2 = .@W2 (200x2), s2/d2 projections. Wave per node.
__global__ __launch_bounds__(256) void node_l1_to_l2(
    const float* __restrict__ agg, const float* __restrict__ b1,
    const float* __restrict__ W2, const float* __restrict__ a2s,
    const float* __restrict__ a2d, float* __restrict__ h2,
    float* __restrict__ s2, float* __restrict__ d2, int n_nodes)
{
    int wave = (int)((blockIdx.x * blockDim.x + threadIdx.x) >> 5);
    int lane = threadIdx.x & 31;
    if (wave >= n_nodes) return;
    const float* ar = agg + (long)wave * HID;
    float r0 = 0.f, r1 = 0.f;
    for (int f = lane; f < HID; f += 32) {
        float v = ar[f] + b1[f];
        v = v > 0.f ? v : 0.f;
        r0 += v * W2[f * OUT_F + 0];
        r1 += v * W2[f * OUT_F + 1];
    }
#pragma unroll
    for (int o = 16; o > 0; o >>= 1) {
        r0 += __shfl_down(r0, o, 32);
        r1 += __shfl_down(r1, o, 32);
    }
    if (lane == 0) {
        h2[wave * 2 + 0] = r0;
        h2[wave * 2 + 1] = r1;
        s2[wave] = r0 * a2s[0] + r1 * a2s[1];
        d2[wave] = r0 * a2d[0] + r1 * a2d[1];
    }
}

__global__ void edge_agg2(const int* __restrict__ src, const int* __restrict__ dst,
                          int E_, int n_nodes, const float* __restrict__ s,
                          const float* __restrict__ d, const float* __restrict__ m,
                          const float* __restrict__ den, const float* __restrict__ h2,
                          float* __restrict__ out)
{
    int e = blockIdx.x * blockDim.x + threadIdx.x;
    int tot = E_ + n_nodes;
    if (e >= tot) return;
    int si = e < E_ ? src[e] : (e - E_);
    int di = e < E_ ? dst[e] : (e - E_);
    float coef = __expf(leaky(s[si] + d[di]) - m[di]) / (den[di] + 1e-16f);
    atomicAdd(out + di * 2 + 0, coef * h2[si * 2 + 0]);
    atomicAdd(out + di * 2 + 1, coef * h2[si * 2 + 1]);
}

__global__ void finalize(float* __restrict__ out, const float* __restrict__ b2, int n2) {
    int i = blockIdx.x * blockDim.x + threadIdx.x;
    if (i >= n2) return;
    float v = out[i] + b2[i & 1];
    out[i] = v > 0.f ? v : 0.f;
}

// ------------------------------ host launcher ------------------------------

extern "C" void kernel_launch(void* const* d_in, const int* in_sizes, int n_in,
                              void* d_out, int out_size, void* d_ws, size_t ws_size,
                              hipStream_t stream) {
    const float* x      = (const float*)d_in[0];
    const int*   eidx   = (const int*)  d_in[1];
    const float* W1     = (const float*)d_in[2];
    const float* a1_src = (const float*)d_in[3];
    const float* a1_dst = (const float*)d_in[4];
    const float* b1     = (const float*)d_in[5];
    const float* W2     = (const float*)d_in[6];
    const float* a2_src = (const float*)d_in[7];
    const float* a2_dst = (const float*)d_in[8];
    const float* b2     = (const float*)d_in[9];
    float* out = (float*)d_out;

    const int N = in_sizes[0] / IN_F;
    const int E = in_sizes[1] / 2;
    const int tot = E + N;
    const int* e_src = eidx;
    const int* e_dst = eidx + E;

    // Workspace carve-out (256B aligned slices).
    char* base = (char*)d_ws;
    size_t off = 0;
    auto alloc = [&](size_t bytes) -> void* {
        void* p = base + off;
        off = (off + bytes + 255) & ~(size_t)255;
        return p;
    };
    float*  h1    = (float*) alloc((size_t)N * HID_P * 4);
    bf16_t* Whi   = (bf16_t*)alloc((size_t)HID_P * IN_F * 2);
    bf16_t* Wlo   = (bf16_t*)alloc((size_t)HID_P * IN_F * 2);
    float*  a1s_p = (float*) alloc((size_t)HID_P * 4);
    float*  a1d_p = (float*) alloc((size_t)HID_P * 4);
    float*  s1    = (float*) alloc((size_t)N * 4);
    float*  d1    = (float*) alloc((size_t)N * 4);
    float*  m1    = (float*) alloc((size_t)N * 4);
    float*  den1  = (float*) alloc((size_t)N * 4);
    float*  agg1  = (float*) alloc((size_t)N * HID * 4);
    float*  h2    = (float*) alloc((size_t)N * OUT_F * 4);
    float*  s2    = (float*) alloc((size_t)N * 4);
    float*  d2    = (float*) alloc((size_t)N * 4);
    float*  m2    = (float*) alloc((size_t)N * 4);
    float*  den2  = (float*) alloc((size_t)N * 4);
    (void)ws_size; (void)n_in; (void)out_size;

    const int B = 256;
    // 0) pack W1 (+ padded a1 vectors) into workspace
    pack_w1<<<(HID_P * IN_F + B - 1) / B, B, 0, stream>>>(W1, a1_src, a1_dst,
                                                          Whi, Wlo, a1s_p, a1d_p);
    // 1) init accumulators / segment-reduction state
    fill_f32<<<2048, B, 0, stream>>>(agg1, (long)N * HID, 0.0f);
    fill_f32<<<256,  B, 0, stream>>>(out,  (long)N * OUT_F, 0.0f);
    fill_f32<<<256,  B, 0, stream>>>(m1,   N, -FLT_MAX);
    fill_f32<<<256,  B, 0, stream>>>(den1, N, 0.0f);
    fill_f32<<<256,  B, 0, stream>>>(m2,   N, -FLT_MAX);
    fill_f32<<<256,  B, 0, stream>>>(den2, N, 0.0f);
    // 2) layer-1 GEMM via WMMA + fused s1/d1 (one wave per 16-row tile)
    {
        int mtiles = N / 16;
        gemm1_wmma<<<(mtiles + 7) / 8, B, 0, stream>>>(x, Whi, Wlo, a1s_p, a1d_p,
                                                       h1, s1, d1, N);
    }
    // 3) layer-1 segment softmax + aggregation
    edge_max   <<<(tot + B - 1) / B, B, 0, stream>>>(e_src, e_dst, E, N, s1, d1, m1);
    edge_expsum<<<(tot + B - 1) / B, B, 0, stream>>>(e_src, e_dst, E, N, s1, d1, m1, den1);
    edge_agg1  <<<(tot + 7) / 8,      B, 0, stream>>>(e_src, e_dst, E, N, s1, d1, m1, den1, h1, agg1);
    // 4) relu(+b1), layer-2 projection h2 = .@W2, s2/d2
    node_l1_to_l2<<<(N + 7) / 8, B, 0, stream>>>(agg1, b1, W2, a2_src, a2_dst, h2, s2, d2, N);
    // 5) layer-2 segment softmax + aggregation directly into d_out
    edge_max   <<<(tot + B - 1) / B, B, 0, stream>>>(e_src, e_dst, E, N, s2, d2, m2);
    edge_expsum<<<(tot + B - 1) / B, B, 0, stream>>>(e_src, e_dst, E, N, s2, d2, m2, den2);
    edge_agg2  <<<(tot + B - 1) / B, B, 0, stream>>>(e_src, e_dst, E, N, s2, d2, m2, den2, h2, out);
    // 6) +b2, relu
    finalize<<<((N * OUT_F) + B - 1) / B, B, 0, stream>>>(out, b2, N * OUT_F);
}
